// VoxtralAudioTokenizer_3393024164129
// MI455X (gfx1250) — matrix operand order repaired and checked by reading, not compiled, not run on hardware
//
#include <hip/hip_runtime.h>
#include <hip/hip_bf16.h>

// ---------------------------------------------------------------------------
// VoxtralAudioTokenizer forward for MI455X (gfx1250, wave32, WMMA).
// GEMMs run in bf16 via v_wmma_f32_16x16x32_bf16 with f32 accumulation.
// Tile staging uses the CDNA5 async global->LDS DMA path when available.
// ---------------------------------------------------------------------------

typedef unsigned short us16;
typedef __attribute__((ext_vector_type(16))) __bf16 v16bf;
typedef __attribute__((ext_vector_type(8)))  __bf16 v8bf;
typedef __attribute__((ext_vector_type(8)))  float  v8f;

#define NH_      8
#define HD_      128
#define DIM_     1024
#define HID_     4096
#define WINDOW_  256
#define AC_DIM_  36
#define NORM_EPS_ 0.01f
#define QK_EPS_   1e-6f

// Staging path selection:
//  1) async global->LDS (GLOBAL_LOAD_ASYNC_TO_LDS_B128 + s_wait_asynccnt)
//  2) Tensor Data Mover (5-arg builtin, ROCm 7.2 / clang-22 only)
//  3) manual global_load_b128 -> ds_store_b128
#if defined(__has_builtin)
#  if __has_builtin(__builtin_amdgcn_global_load_async_to_lds_b128) && \
      __has_builtin(__builtin_amdgcn_s_wait_asynccnt)
#    define USE_ASYNC 1
#  elif __has_builtin(__builtin_amdgcn_tensor_load_to_lds) && (__clang_major__ <= 22)
#    define USE_TDM 1
#  endif
#endif

typedef __attribute__((ext_vector_type(4))) unsigned int u32x4;
typedef __attribute__((ext_vector_type(8))) int i32x8;
typedef __attribute__((ext_vector_type(4))) int i32x4;
typedef __attribute__((address_space(3))) unsigned short as3_us16;
// async-DMA builtin takes int4* pointers: global (AS1) src, LDS (AS3) dst
typedef __attribute__((address_space(1))) i32x4 as1_i32x4;
typedef __attribute__((address_space(3))) i32x4 as3_i32x4;

__device__ __forceinline__ unsigned lds_byte_off(const us16* p) {
  return (unsigned)(unsigned long long)(const as3_us16*)p;
}

__device__ __forceinline__ us16 f2bf(float f) {
  unsigned int u = __float_as_uint(f);
  u += 0x7fffu + ((u >> 16) & 1u);          // round-to-nearest-even
  return (us16)(u >> 16);
}

// ---------------------------------------------------------------------------
// f32 -> bf16 cast
// ---------------------------------------------------------------------------
__global__ void k_cvt(const float* __restrict__ s, us16* __restrict__ d, int n) {
  int i = blockIdx.x * blockDim.x + threadIdx.x;
  if (i < n) d[i] = f2bf(s[i]);
}

// ---------------------------------------------------------------------------
// Embedding + acoustic-code concat: x0 (B, 256+36, T)
// ---------------------------------------------------------------------------
__global__ void k_embed(const int* __restrict__ codes, const float* __restrict__ emb,
                        const float* __restrict__ usage, float* __restrict__ x0,
                        int Bn, int T) {
  int idx = blockIdx.x * blockDim.x + threadIdx.x;
  int C = 256 + AC_DIM_;
  int total = Bn * C * T;
  if (idx >= total) return;
  int t = idx % T, c = (idx / T) % C, b = idx / (T * C);
  if (c < 256) {
    int code = codes[((size_t)b * (1 + AC_DIM_) + 0) * T + t];
    float u = usage[code]; u = (u < 1e-5f) ? 1e-5f : u;
    x0[idx] = emb[(size_t)code * 256 + c] / u;
  } else {
    int a = codes[((size_t)b * (1 + AC_DIM_) + 1 + (c - 256)) * T + t];
    x0[idx] = (float)a * 0.1f - 1.0f;   // 2/(21-1)
  }
}

// ---------------------------------------------------------------------------
// Stride-1 causal conv with reflect-left pad (all forward convs here are s=1)
// ---------------------------------------------------------------------------
__global__ void k_conv(const float* __restrict__ x, const float* __restrict__ w,
                       const float* __restrict__ bias, float* __restrict__ y,
                       int Bn, int Cin, int Cout, int L, int Kk) {
  int idx = blockIdx.x * blockDim.x + threadIdx.x;
  int total = Bn * Cout * L;
  if (idx >= total) return;
  int t = idx % L, oc = (idx / L) % Cout, b = idx / (L * Cout);
  float acc = bias[oc];
  const float* xb = x + (size_t)b * Cin * L;
  const float* wr = w + (size_t)oc * Cin * Kk;
  for (int i = 0; i < Cin; ++i) {
    const float* xi = xb + (size_t)i * L;
    const float* wi = wr + (size_t)i * Kk;
    for (int k = 0; k < Kk; ++k) {
      int p = t - (Kk - 1) + k;
      p = p < 0 ? -p : p;                  // reflect (no edge repeat)
      acc += xi[p] * wi[k];
    }
  }
  y[idx] = acc;
}

// ---------------------------------------------------------------------------
// Causal conv-transpose, weights (I,O,K), output trimmed to Lin*stride
// ---------------------------------------------------------------------------
__global__ void k_convt(const float* __restrict__ x, const float* __restrict__ w,
                        const float* __restrict__ bias, float* __restrict__ y,
                        int Bn, int C, int Lin, int Kk, int stride) {
  int Lout = Lin * stride;
  int idx = blockIdx.x * blockDim.x + threadIdx.x;
  int total = Bn * C * Lout;
  if (idx >= total) return;
  int t = idx % Lout, oc = (idx / Lout) % C, b = idx / (Lout * C);
  float acc = bias[oc];
  const float* xb = x + (size_t)b * C * Lin;
  for (int k = 0; k < Kk; ++k) {
    int tp = t - k;
    if (tp < 0 || (tp % stride)) continue;
    int s = tp / stride;
    if (s >= Lin) continue;
    for (int i = 0; i < C; ++i)
      acc += xb[(size_t)i * Lin + s] * w[((size_t)i * C + oc) * Kk + k];
  }
  y[idx] = acc;
}

// (B,C,L) -> (B*L, C) and inverse
__global__ void k_tr_cl2lc(const float* __restrict__ x, float* __restrict__ y,
                           int Bn, int C, int L) {
  int idx = blockIdx.x * blockDim.x + threadIdx.x;
  int total = Bn * C * L;
  if (idx >= total) return;
  int t = idx % L, c = (idx / L) % C, b = idx / (L * C);
  y[((size_t)(b * L + t)) * C + c] = x[idx];
}
__global__ void k_tr_lc2cl(const float* __restrict__ x, float* __restrict__ y,
                           int Bn, int C, int L) {
  int idx = blockIdx.x * blockDim.x + threadIdx.x;
  int total = Bn * C * L;
  if (idx >= total) return;
  int c = idx % C, t = (idx / C) % L, b = idx / (C * L);
  y[((size_t)(b * C + c)) * L + t] = x[idx];
}

// ---------------------------------------------------------------------------
// RMS norm over a row of C, f32 in -> bf16 out
// ---------------------------------------------------------------------------
__global__ __launch_bounds__(256)
void k_rms(const float* __restrict__ x, const float* __restrict__ w,
           us16* __restrict__ out, int C, float eps) {
  int row = blockIdx.x;
  const float* xr = x + (size_t)row * C;
  __shared__ float red[256];
  float s = 0.f;
  for (int c = threadIdx.x; c < C; c += 256) { float v = xr[c]; s += v * v; }
  red[threadIdx.x] = s; __syncthreads();
  for (int o = 128; o > 0; o >>= 1) {
    if ((int)threadIdx.x < o) red[threadIdx.x] += red[threadIdx.x + o];
    __syncthreads();
  }
  float r = rsqrtf(red[0] / (float)C + eps);
  for (int c = threadIdx.x; c < C; c += 256)
    out[(size_t)row * C + c] = f2bf(xr[c] * r * w[c]);
}

// ---------------------------------------------------------------------------
// Per-head QK RMS norm + pack bf16 Q,K (b,h,t,d) and V transposed (b,h,d,t)
// ---------------------------------------------------------------------------
__global__ __launch_bounds__(128)
void k_qknorm(const float* __restrict__ qkv, const float* __restrict__ qn,
              const float* __restrict__ kn, us16* __restrict__ Q,
              us16* __restrict__ Kc, us16* __restrict__ Vt, int T) {
  int h = blockIdx.x & (NH_ - 1);
  int bt = blockIdx.x >> 3;
  int t = bt % T, b = bt / T;
  int d = threadIdx.x;
  const float* base = qkv + (size_t)bt * (3 * DIM_);
  __shared__ float red[128];

  float qv = base[h * HD_ + d];
  red[d] = qv * qv; __syncthreads();
  for (int o = 64; o > 0; o >>= 1) { if (d < o) red[d] += red[d + o]; __syncthreads(); }
  float qr = rsqrtf(red[0] / (float)HD_ + QK_EPS_);
  __syncthreads();

  float kv = base[DIM_ + h * HD_ + d];
  red[d] = kv * kv; __syncthreads();
  for (int o = 64; o > 0; o >>= 1) { if (d < o) red[d] += red[d + o]; __syncthreads(); }
  float kr = rsqrtf(red[0] / (float)HD_ + QK_EPS_);

  size_t qi = ((size_t)(b * NH_ + h) * T + t) * HD_ + d;
  Q[qi]  = f2bf(qv * qr * qn[d]);
  Kc[qi] = f2bf(kv * kr * kn[d]);
  float vv = base[2 * DIM_ + h * HD_ + d];
  Vt[((size_t)(b * NH_ + h) * HD_ + d) * T + t] = f2bf(vv);
}

// ---------------------------------------------------------------------------
// bf16 WMMA GEMM: Y[M,N] = A[M,K] * W[N,K]^T, 64x64 tile / 4 waves, K step 32,
// double-buffered LDS staged via async global->LDS DMA (ASYNCcnt) if present.
// mode 0: store f32            mode 2: outF = resid + acc*scale[col]
// mode 3: outH(bf16) = silu(aux) * acc   (SwiGLU fusion)
// ---------------------------------------------------------------------------
__global__ __launch_bounds__(128)
void k_gemm(const us16* __restrict__ A, const us16* __restrict__ W,
            int M, int N, int K, float* __restrict__ outF, us16* __restrict__ outH,
            const float* __restrict__ resid, const float* __restrict__ scale,
            const float* __restrict__ aux, int mode) {
  __shared__ __align__(32) us16 lA[2][64 * 32];
  __shared__ __align__(32) us16 lW[2][64 * 32];
  int tid = threadIdx.x, lane = tid & 31, wave = tid >> 5;
  int n0 = blockIdx.x * 64, m0 = blockIdx.y * 64;

  v8f acc[4];
  for (int j = 0; j < 4; ++j)
    for (int e = 0; e < 8; ++e) acc[j][e] = 0.f;

  int r = tid >> 1, c0 = (tid & 1) * 16;
  const us16* Asrc = A + (size_t)(m0 + r) * K + c0;
  const us16* Wsrc = W + (size_t)(n0 + r) * K + c0;
  int mfr = lane & 15, half = lane >> 4;

  auto stage = [&](int k0, int buf) {
#if defined(USE_ASYNC)
    // CDNA5 async DMA: global -> LDS without touching VGPRs. INST_OFFSET is
    // added to both the LDS and global address (ISA 10.x), which matches our
    // +16B second chunk on both sides.
    __builtin_amdgcn_global_load_async_to_lds_b128(
        (as1_i32x4*)(Asrc + k0), (as3_i32x4*)&lA[buf][r * 32 + c0], 0, 0);
    __builtin_amdgcn_global_load_async_to_lds_b128(
        (as1_i32x4*)(Asrc + k0), (as3_i32x4*)&lA[buf][r * 32 + c0], 16, 0);
    __builtin_amdgcn_global_load_async_to_lds_b128(
        (as1_i32x4*)(Wsrc + k0), (as3_i32x4*)&lW[buf][r * 32 + c0], 0, 0);
    __builtin_amdgcn_global_load_async_to_lds_b128(
        (as1_i32x4*)(Wsrc + k0), (as3_i32x4*)&lW[buf][r * 32 + c0], 16, 0);
#elif defined(USE_TDM)
    // A tile: manual vectorized staging
    const uint4* sa = (const uint4*)(Asrc + k0);
    uint4 a0 = sa[0], a1 = sa[1];
    *(uint4*)&lA[buf][r * 32 + c0]     = a0;
    *(uint4*)&lA[buf][r * 32 + c0 + 8] = a1;
    if (wave == 0) {
      // W tile via TDM: 2-D D#, 64 rows x 32 cols of row-major (N,K) bf16
      unsigned long long ga =
          (unsigned long long)(const void*)W + ((size_t)n0 * K + k0) * 2ull;
      unsigned lds = lds_byte_off(&lW[buf][0]);
      u32x4 g0; i32x8 g1; i32x4 g2, g3;
      g0[0] = 1u;                                            // count=1
      g0[1] = lds;                                           // lds_addr
      g0[2] = (unsigned)(ga & 0xffffffffull);                // global_addr lo
      g0[3] = (unsigned)((ga >> 32) & 0x01ffffffull) | (2u << 30);  // hi | type=2
      g1[0] = (int)(1u << 16);                               // data_size = 2B
      g1[1] = (int)(((unsigned)K & 0xffffu) << 16);          // tensor_dim0 lo
      g1[2] = (int)((((unsigned)K >> 16) & 0xffffu) |
                    (((unsigned)N & 0xffffu) << 16));        // dim0 hi | dim1 lo
      g1[3] = (int)((((unsigned)N >> 16) & 0xffffu) |
                    (32u << 16));                            // dim1 hi | tile_dim0
      g1[4] = (int)64u;                                      // tile_dim1
      g1[5] = (int)K;                                        // dim0_stride lo
      g1[6] = 0; g1[7] = 0;
      g2[0] = g2[1] = g2[2] = g2[3] = 0;
      g3[0] = g3[1] = g3[2] = g3[3] = 0;
      __builtin_amdgcn_tensor_load_to_lds(g0, g1, g2, g3, 0);
    }
#else
    const uint4* sa = (const uint4*)(Asrc + k0);
    const uint4* sw = (const uint4*)(Wsrc + k0);
    uint4 a0 = sa[0], a1 = sa[1], w0 = sw[0], w1 = sw[1];
    *(uint4*)&lA[buf][r * 32 + c0]     = a0;
    *(uint4*)&lA[buf][r * 32 + c0 + 8] = a1;
    *(uint4*)&lW[buf][r * 32 + c0]     = w0;
    *(uint4*)&lW[buf][r * 32 + c0 + 8] = w1;
#endif
  };

  stage(0, 0);
  int nk = K / 32;
  for (int ki = 0; ki < nk; ++ki) {
    int cur = ki & 1;
#if defined(USE_ASYNC)
    __builtin_amdgcn_s_wait_asynccnt(0);
#elif defined(USE_TDM)
    if (wave == 0) __builtin_amdgcn_s_wait_tensorcnt(0);
#endif
    __syncthreads();
    if (ki + 1 < nk) {
      stage((ki + 1) * 32, cur ^ 1);
      __builtin_prefetch(Asrc + (size_t)(ki + 2) * 32, 0, 3);
#if !defined(USE_TDM)
      __builtin_prefetch(Wsrc + (size_t)(ki + 2) * 32, 0, 3);
#endif
    }

    // A fragment (16x32 bf16): elems 0-7 <- K[half*8..+8), 8-15 <- K[16+half*8..+8)
    const us16* ap = &lA[cur][(wave * 16 + mfr) * 32 + half * 8];
    v16bf a;
    ((v8bf*)&a)[0] = *(const v8bf*)ap;
    ((v8bf*)&a)[1] = *(const v8bf*)(ap + 16);

    for (int j = 0; j < 4; ++j) {
      // B fragment (32x16): elems 0-7 <- K[half*8..+8), 8-15 <- K[16+half*8..+8) of W row (n)
      const us16* bp = &lW[cur][(j * 16 + mfr) * 32 + half * 8];
      v16bf b;
      ((v8bf*)&b)[0] = *(const v8bf*)bp;
      ((v8bf*)&b)[1] = *(const v8bf*)(bp + 16);
      acc[j] = __builtin_amdgcn_wmma_f32_16x16x32_bf16(
          false, a, false, b, (short)0, acc[j], false, false);
    }
  }

  for (int j = 0; j < 4; ++j) {
    int col = n0 + j * 16 + mfr;
    for (int v = 0; v < 8; ++v) {
      int row = m0 + wave * 16 + v + 8 * half;
      size_t idx = (size_t)row * N + col;
      float val = acc[j][v];
      if (mode == 0) {
        outF[idx] = val;
      } else if (mode == 2) {
        outF[idx] = resid[idx] + val * scale[col];
      } else {  // mode 3: SwiGLU
        float g = aux[idx];
        float s = g / (1.f + __expf(-g));
        outH[idx] = f2bf(s * val);
      }
    }
  }
}

// ---------------------------------------------------------------------------
// Windowed causal flash-attention. One wave per (16-query tile, head, batch).
// S = Q K^T / sqrt(HD) via WMMA; online softmax; Y += P V via WMMA (V^T feed).
// Output bf16 rows (b*T+t, h*128+d) ready for the WO GEMM.
// ---------------------------------------------------------------------------
__global__ __launch_bounds__(32)
void k_attn(const us16* __restrict__ Q, const us16* __restrict__ Kb,
            const us16* __restrict__ Vt, us16* __restrict__ out, int T) {
  int qt = blockIdx.x, h = blockIdx.y, b = blockIdx.z;
  int lane = threadIdx.x, mfr = lane & 15, hl = lane >> 4;
  int q0 = qt * 16;
  const us16* Qh = Q  + (size_t)(b * NH_ + h) * T * HD_;
  const us16* Kh = Kb + (size_t)(b * NH_ + h) * T * HD_;
  const us16* Vh = Vt + (size_t)(b * NH_ + h) * HD_ * T;
  __shared__ __align__(32) us16 lP[16 * 32];

  // Q A-fragments for the 4 K-chunks of HD=128
  v16bf qf[4];
  for (int kc = 0; kc < 4; ++kc) {
    const us16* qp = Qh + (size_t)(q0 + mfr) * HD_ + kc * 32 + hl * 8;
    ((v8bf*)&qf[kc])[0] = *(const v8bf*)qp;
    ((v8bf*)&qf[kc])[1] = *(const v8bf*)(qp + 16);
  }

  v8f y[8];
  for (int dn = 0; dn < 8; ++dn)
    for (int e = 0; e < 8; ++e) y[dn][e] = 0.f;
  float rm[8], rl[8];
  for (int v = 0; v < 8; ++v) { rm[v] = -3.0e38f; rl[v] = 0.f; }

  int jmin = q0 + 15 - (WINDOW_ - 1); if (jmin < 0) jmin = 0;
  int kb0 = jmin & ~31;
  const float sc = 0.08838834764831845f;  // 1/sqrt(128)

  for (int kb = kb0; kb <= q0 + 15; kb += 32) {
    v8f s0, s1;
    for (int e = 0; e < 8; ++e) { s0[e] = 0.f; s1[e] = 0.f; }
    for (int kc = 0; kc < 4; ++kc) {
      const us16* kp0 = Kh + (size_t)(kb + mfr)      * HD_ + kc * 32 + hl * 8;
      const us16* kp1 = Kh + (size_t)(kb + 16 + mfr) * HD_ + kc * 32 + hl * 8;
      v16bf b0, b1;
      ((v8bf*)&b0)[0] = *(const v8bf*)kp0; ((v8bf*)&b0)[1] = *(const v8bf*)(kp0 + 16);
      ((v8bf*)&b1)[0] = *(const v8bf*)kp1; ((v8bf*)&b1)[1] = *(const v8bf*)(kp1 + 16);
      s0 = __builtin_amdgcn_wmma_f32_16x16x32_bf16(false, qf[kc], false, b0, (short)0, s0, false, false);
      s1 = __builtin_amdgcn_wmma_f32_16x16x32_bf16(false, qf[kc], false, b1, (short)0, s1, false, false);
    }
    // mask + scale
    for (int v = 0; v < 8; ++v) {
      int i = q0 + v + 8 * hl;
      int j0 = kb + mfr, j1 = kb + 16 + mfr;
      bool ok0 = (j0 <= i) && (i - j0 < WINDOW_);
      bool ok1 = (j1 <= i) && (i - j1 < WINDOW_);
      s0[v] = ok0 ? s0[v] * sc : -3.0e38f;
      s1[v] = ok1 ? s1[v] * sc : -3.0e38f;
    }
    float p0[8], p1[8], alpha[8];
    for (int v = 0; v < 8; ++v) {
      float mx = fmaxf(s0[v], s1[v]);
      mx = fmaxf(mx, __shfl_xor(mx, 1));
      mx = fmaxf(mx, __shfl_xor(mx, 2));
      mx = fmaxf(mx, __shfl_xor(mx, 4));
      mx = fmaxf(mx, __shfl_xor(mx, 8));
      float nm = fmaxf(rm[v], mx);
      float safe = fmaxf(nm, -1e30f);      // all-masked tile -> p = 0, alpha = 0
      alpha[v] = __expf(rm[v] - safe);
      rm[v] = nm;
      p0[v] = __expf(s0[v] - safe);
      p1[v] = __expf(s1[v] - safe);
      float ps = p0[v] + p1[v];
      ps += __shfl_xor(ps, 1); ps += __shfl_xor(ps, 2);
      ps += __shfl_xor(ps, 4); ps += __shfl_xor(ps, 8);
      rl[v] = rl[v] * alpha[v] + ps;
    }
    for (int dn = 0; dn < 8; ++dn)
      for (int v = 0; v < 8; ++v) y[dn][v] *= alpha[v];

    // transpose P through LDS into A-fragment layout (bf16 16x32)
    __syncthreads();
    for (int v = 0; v < 8; ++v) {
      lP[(v + 8 * hl) * 32 + mfr]      = f2bf(p0[v]);
      lP[(v + 8 * hl) * 32 + 16 + mfr] = f2bf(p1[v]);
    }
    __syncthreads();
    v16bf pf;
    const us16* pp = &lP[mfr * 32 + hl * 8];
    ((v8bf*)&pf)[0] = *(const v8bf*)pp;
    ((v8bf*)&pf)[1] = *(const v8bf*)(pp + 16);

    for (int dn = 0; dn < 8; ++dn) {
      // V^T row d = dn*16+n, keys kb + hl*8..+8 and kb + 16 + hl*8..+8
      const us16* vp = Vh + (size_t)(dn * 16 + mfr) * T + kb + hl * 8;
      v16bf vb;
      ((v8bf*)&vb)[0] = *(const v8bf*)vp;
      ((v8bf*)&vb)[1] = *(const v8bf*)(vp + 16);
      y[dn] = __builtin_amdgcn_wmma_f32_16x16x32_bf16(false, pf, false, vb, (short)0, y[dn], false, false);
    }
  }

  for (int dn = 0; dn < 8; ++dn)
    for (int v = 0; v < 8; ++v) {
      int t = q0 + v + 8 * hl;
      out[((size_t)(b * T + t)) * (NH_ * HD_) + h * HD_ + dn * 16 + mfr] =
          f2bf(y[dn][v] / rl[v]);
    }
}

// ---------------------------------------------------------------------------
// Final (B,P,L) -> (B,1,L*P) reshape
// ---------------------------------------------------------------------------
__global__ void k_outre(const float* __restrict__ y, float* __restrict__ out,
                        int Bn, int P, int L) {
  int idx = blockIdx.x * blockDim.x + threadIdx.x;
  int total = Bn * P * L;
  if (idx >= total) return;
  int t = idx % L, p = (idx / L) % P, b = idx / (L * P);
  out[(size_t)b * L * P + (size_t)t * P + p] = y[idx];
}

// ---------------------------------------------------------------------------
// Host orchestration
// ---------------------------------------------------------------------------
extern "C" void kernel_launch(void* const* d_in, const int* in_sizes, int n_in,
                              void* d_out, int out_size, void* d_ws, size_t ws_size,
                              hipStream_t stream) {
  (void)in_sizes; (void)n_in; (void)out_size; (void)ws_size;
  // Input flattening assumed JAX-pytree (sorted dict keys):
  //   0: codes
  //   params: blocks(4x2 layers x 13 sorted leaves), convs(4x{b,w}),
  //           out_proj{b,w}, patch_proj{b,w}, sem_emb_sum, sem_usage
  // layer leaves: 0 attn_norm 1 attn_scale 2 ffn_norm 3 ffn_scale 4 k_norm
  //               5 q_norm 6 w1 7 w2 8 w3 9 wk 10 wo 11 wq 12 wv
  const int* codes = (const int*)d_in[0];
  auto F = [&](int i) { return (const float*)d_in[i]; };
  const int CONVS_BASE = 1 + 8 * 13;     // 105
  const int OUTP_B = 113, OUTP_W = 114, PATCH_B = 115, PATCH_W = 116;
  const int SEM_EMB = 117, SEM_USE = 118;

  // bump allocator over ws (256-B aligned)
  char* wp = (char*)d_ws;
  size_t off = 0;
  auto alloc = [&](size_t bytes) -> void* {
    off = (off + 255) & ~(size_t)255;
    void* p = wp + off;
    off += bytes;
    return p;
  };
  const int Bn = 2, Lmax = 1536, Mmax = 2 * Lmax;
  float* xA      = (float*)alloc((size_t)Bn * DIM_ * Lmax * 4);
  float* xB      = (float*)alloc((size_t)Bn * DIM_ * Lmax * 4);
  float* hbuf    = (float*)alloc((size_t)Mmax * DIM_ * 4);
  float* qkv     = (float*)alloc((size_t)Mmax * 3 * DIM_ * 4);
  us16*  Qb      = (us16*) alloc((size_t)Mmax * DIM_ * 2);
  us16*  Kbuf    = (us16*) alloc((size_t)Mmax * DIM_ * 2);
  us16*  Vt      = (us16*) alloc((size_t)Mmax * DIM_ * 2);
  us16*  Xbf     = (us16*) alloc((size_t)Mmax * DIM_ * 2);
  us16*  XbfBig  = (us16*) alloc((size_t)Mmax * HID_ * 2);
  float* buf1    = (float*)alloc((size_t)Mmax * HID_ * 4);
  us16*  Wscr    = (us16*) alloc((size_t)HID_ * DIM_ * 2);
  float* x0f     = (float*)alloc((size_t)Bn * (256 + AC_DIM_) * 192 * 4);

  auto cvt = [&](const float* s, us16* d, int n) {
    k_cvt<<<(n + 255) / 256, 256, 0, stream>>>(s, d, n);
  };
  auto gemm = [&](const us16* A, const us16* W, int M, int N, int K,
                  float* oF, us16* oH, const float* resid, const float* scl,
                  const float* aux, int mode) {
    dim3 g(N / 64, M / 64);
    k_gemm<<<g, 128, 0, stream>>>(A, W, M, N, K, oF, oH, resid, scl, aux, mode);
  };

  auto run_block = [&](int li, int T) {
    int M = Bn * T;
    int base = 1 + li * 13;
    // --- attention ---
    cvt(F(base + 11), Wscr,                 DIM_ * DIM_);   // wq
    cvt(F(base +  9), Wscr + DIM_ * DIM_,   DIM_ * DIM_);   // wk
    cvt(F(base + 12), Wscr + 2 * DIM_ * DIM_, DIM_ * DIM_); // wv
    k_rms<<<M, 256, 0, stream>>>(hbuf, F(base + 0), Xbf, DIM_, NORM_EPS_);
    gemm(Xbf, Wscr, M, 3 * DIM_, DIM_, qkv, nullptr, nullptr, nullptr, nullptr, 0);
    k_qknorm<<<M * NH_, 128, 0, stream>>>(qkv, F(base + 5), F(base + 4), Qb, Kbuf, Vt, T);
    dim3 ga(T / 16, NH_, Bn);
    k_attn<<<ga, 32, 0, stream>>>(Qb, Kbuf, Vt, Xbf, T);
    cvt(F(base + 10), Wscr, DIM_ * DIM_);                   // wo
    gemm(Xbf, Wscr, M, DIM_, DIM_, hbuf, nullptr, hbuf, F(base + 1), nullptr, 2);
    // --- FFN (SwiGLU) ---
    k_rms<<<M, 256, 0, stream>>>(hbuf, F(base + 2), Xbf, DIM_, NORM_EPS_);
    cvt(F(base + 6), Wscr, HID_ * DIM_);                    // w1
    gemm(Xbf, Wscr, M, HID_, DIM_, buf1, nullptr, nullptr, nullptr, nullptr, 0);
    cvt(F(base + 8), Wscr, HID_ * DIM_);                    // w3
    gemm(Xbf, Wscr, M, HID_, DIM_, nullptr, XbfBig, nullptr, nullptr, buf1, 3);
    cvt(F(base + 7), Wscr, DIM_ * HID_);                    // w2
    gemm(XbfBig, Wscr, M, DIM_, HID_, hbuf, nullptr, hbuf, F(base + 3), nullptr, 2);
  };

  // --- front end ---
  {
    int T = 192, C0 = 256 + AC_DIM_;
    int n = Bn * C0 * T;
    k_embed<<<(n + 255) / 256, 256, 0, stream>>>(codes, F(SEM_EMB), F(SEM_USE), x0f, Bn, T);
    int m = Bn * DIM_ * T;
    k_conv<<<(m + 255) / 256, 256, 0, stream>>>(x0f, F(PATCH_W), F(PATCH_B), xA,
                                                Bn, C0, DIM_, T, 7);
  }

  // --- 4 stages ---
  int L = 192;
  const int KER[4] = {3, 4, 4, 4};
  for (int s = 0; s < 4; ++s) {
    const float* cw = F(CONVS_BASE + 2 * s + 1);
    const float* cb = F(CONVS_BASE + 2 * s + 0);
    if (s == 0) {
      int m = Bn * DIM_ * L;
      k_conv<<<(m + 255) / 256, 256, 0, stream>>>(xA, cw, cb, xB, Bn, DIM_, DIM_, L, KER[0]);
    } else {
      int m = Bn * DIM_ * L * 2;
      k_convt<<<(m + 255) / 256, 256, 0, stream>>>(xA, cw, cb, xB, Bn, DIM_, L, KER[s], 2);
      L *= 2;
    }
    int n = Bn * DIM_ * L;
    k_tr_cl2lc<<<(n + 255) / 256, 256, 0, stream>>>(xB, hbuf, Bn, DIM_, L);
    run_block(s * 2 + 0, L);
    run_block(s * 2 + 1, L);
    k_tr_lc2cl<<<(n + 255) / 256, 256, 0, stream>>>(hbuf, xA, Bn, DIM_, L);
  }

  // --- output head ---
  {
    int P = 240;
    int m = Bn * P * L;
    k_conv<<<(m + 255) / 256, 256, 0, stream>>>(xA, F(OUTP_W), F(OUTP_B), xB,
                                                Bn, DIM_, P, L, 7);
    k_outre<<<(m + 255) / 256, 256, 0, stream>>>(xB, (float*)d_out, Bn, P, L);
  }
}
